// GVPConvLayer_29858612642353
// MI455X (gfx1250) — matrix operand order repaired and checked by reading, not compiled, not run on hardware
//
#include <hip/hip_runtime.h>
#include <math.h>

// ---------------- types ----------------
typedef _Float16 h8_t  __attribute__((ext_vector_type(8)));
typedef _Float16 h16_t __attribute__((ext_vector_type(16)));
typedef float    f8_t  __attribute__((ext_vector_type(8)));

// ---------------- problem dims ----------------
constexpr int kNS  = 128;            // node scalar channels
constexpr int kNV  = 32;             // node vector channels
constexpr int kES  = 32;             // edge scalar channels
constexpr int kSI0 = 2 * kNS + kES;  // 288 concat scalar in
constexpr int kVI0 = 2 * kNV + 1;    // 65 concat vector in
constexpr int kK0  = 384;            // 353 padded to mult of 32
constexpr int kK1  = 160;            // 128 + 32
constexpr int kHS  = 512;            // FF hidden scalar
constexpr int kHV  = 64;             // FF hidden vector
constexpr int kKF0 = 192;            // 128 + 64
constexpr int kKF1 = 576;            // 512 + 64
constexpr float kEps   = 1e-8f;
constexpr float kLnEps = 1e-5f;

// f16 weight pool offsets (in halves)
constexpr int OFF_W0 = 0;                       // msg0_wsw [128][384]
constexpr int OFF_W1 = OFF_W0 + 128 * kK0;      // msg1_wsw [128][160]
constexpr int OFF_W2 = OFF_W1 + 128 * kK1;      // msg2_wsw [128][160]
constexpr int OFF_F0 = OFF_W2 + 128 * kK1;      // ff0_wsw  [512][192]
constexpr int OFF_F1 = OFF_F0 + kHS * kKF0;     // ff1_wsw  [128][576]
constexpr int WTOT   = OFF_F1 + 128 * kKF1;     // 262144 halves = 512 KB

// ---------------- WMMA helpers ----------------
__device__ __forceinline__ f8_t wmma16(h16_t a, h16_t b, f8_t c) {
  // D = A(16x32 f16) * B(32x16 f16) + C(16x16 f32)
  return __builtin_amdgcn_wmma_f32_16x16x32_f16(false, a, false, b, (short)0, c,
                                                false, false);
}

// A fragment: row-major [16][stride] f16. lane<16: M=lane, K={k0..k0+7, k0+16..k0+23}
//             lane>=16: M=lane-16,  K={k0+8..k0+15, k0+24..k0+31}
__device__ __forceinline__ h16_t fragA(const _Float16* base, int stride, int k0,
                                       int lane) {
  const int r  = lane & 15;
  const int kg = (lane >> 4) << 3;  // 0 or 8
  const _Float16* p = base + r * stride + k0 + kg;
  h8_t lo = *(const h8_t*)p;
  h8_t hi = *(const h8_t*)(p + 16);
  h16_t f;
#pragma unroll
  for (int i = 0; i < 8; ++i) { f[i] = lo[i]; f[i + 8] = hi[i]; }
  return f;
}

// B fragment: weight rows are output channels, K contiguous.
// lane<16: N=lane, K={k0..k0+7, k0+16..k0+23} ; lane>=16: N=lane-16, K={k0+8..k0+15, k0+24..k0+31}
__device__ __forceinline__ h16_t fragB(const _Float16* base, int stride, int k0,
                                       int lane) {
  const int n  = lane & 15;
  const int kg = (lane >> 4) << 3;  // 0 or 8
  const _Float16* p = base + n * stride + k0 + kg;
  h8_t lo = *(const h8_t*)p;
  h8_t hi = *(const h8_t*)(p + 16);
  h16_t f;
#pragma unroll
  for (int i = 0; i < 8; ++i) { f[i] = lo[i]; f[i + 8] = hi[i]; }
  return f;
}

__device__ __forceinline__ float sigmoidf_(float x) {
  return 1.f / (1.f + __expf(-x));
}

// ---------------- kernel 0: weight f32->f16 repack ----------------
__global__ __launch_bounds__(256) void convert_weights_kernel(
    const float* w0, const float* w1, const float* w2, const float* f0,
    const float* f1, _Float16* out) {
  int t = blockIdx.x * 256 + threadIdx.x;
  if (t >= WTOT) return;
  float v;
  if (t < OFF_W1) {
    int r = t / kK0, k = t % kK0;
    v = (k < 353) ? w0[r * 353 + k] : 0.f;     // pad K 353..383 with 0
  } else if (t < OFF_W2) {
    v = w1[t - OFF_W1];
  } else if (t < OFF_F0) {
    v = w2[t - OFF_W2];
  } else if (t < OFF_F1) {
    v = f0[t - OFF_F0];
  } else {
    v = f1[t - OFF_F1];
  }
  out[t] = (_Float16)v;
}

// ---------------- kernel 1: per-edge messages + atomic aggregation ----------
__global__ __launch_bounds__(256) void edge_msg_kernel(
    const float* __restrict__ node_s, const float* __restrict__ node_v,
    const float* __restrict__ edge_s, const float* __restrict__ edge_v,
    const int* __restrict__ edge_index, int Ne,
    const float* __restrict__ wh0, const float* __restrict__ wsb0,
    const float* __restrict__ wv0, const float* __restrict__ wh1,
    const float* __restrict__ wsb1, const float* __restrict__ wv1,
    const float* __restrict__ wh2, const float* __restrict__ wsb2,
    const float* __restrict__ wv2, const _Float16* __restrict__ W0,
    const _Float16* __restrict__ W1, const _Float16* __restrict__ W2,
    float* __restrict__ agg_s, float* __restrict__ agg_v,
    float* __restrict__ cnt) {
  __shared__ __align__(16) _Float16 feat0[16][kK0];
  __shared__ __align__(16) _Float16 feat1[16][kK1];
  __shared__ __align__(16) _Float16 feat2[16][kK1];
  __shared__ float mv0[16][3][66];
  __shared__ float vh0[16][3][66];
  __shared__ float mv1[16][3][33];
  __shared__ float vh1[16][3][33];
  __shared__ float mv2[16][3][33];
  __shared__ float vh2[16][3][33];
  __shared__ int sidx[16], didx[16];

  const int tid  = threadIdx.x;
  const int lane = tid & 31;
  const int wid  = tid >> 5;  // 8 waves
  const int e0   = blockIdx.x * 16;

  if (tid < 16) {
    sidx[tid] = edge_index[e0 + tid];       // src (x_j)
    didx[tid] = edge_index[Ne + e0 + tid];  // dst (x_i)
  }
  // zero the K padding of feat0 (cols 353..383)
  for (int t = tid; t < 16 * (kK0 - 353); t += 256) {
    int r = t / (kK0 - 353), k = 353 + t % (kK0 - 353);
    feat0[r][k] = (_Float16)0.f;
  }
  __syncthreads();

  // gather scalar concat [s_j | e_s | s_i] -> feat0 (f16)
  for (int t = tid; t < 16 * kSI0; t += 256) {
    int r = t / kSI0, c = t % kSI0;
    float val;
    if (c < kNS)            val = node_s[sidx[r] * kNS + c];
    else if (c < kNS + kES) val = edge_s[(e0 + r) * kES + (c - kNS)];
    else                    val = node_s[didx[r] * kNS + (c - kNS - kES)];
    feat0[r][c] = (_Float16)val;
  }
  // gather vector concat [v_j | e_v | v_i] -> mv0[e][dim][chan] (f32)
  for (int t = tid; t < 16 * kVI0 * 3; t += 256) {
    int r = t / (kVI0 * 3), rem = t % (kVI0 * 3), c = rem / 3, d = rem % 3;
    float val;
    if (c < kNV)          val = node_v[(sidx[r] * kNV + c) * 3 + d];
    else if (c == kNV)    val = edge_v[(e0 + r) * 3 + d];
    else                  val = node_v[(didx[r] * kNV + (c - kNV - 1)) * 3 + d];
    mv0[r][d][c] = val;
  }
  __syncthreads();

  // ---- GVP0 vector hidden: vh0 = wh0 (65x65) applied per dim; vn -> feat0
  for (int p = tid; p < 16 * kVI0; p += 256) {
    int r = p / kVI0, h = p % kVI0;
    float a0 = 0.f, a1 = 0.f, a2 = 0.f;
    for (int c = 0; c < kVI0; ++c) {
      float w = wh0[h * kVI0 + c];
      a0 += mv0[r][0][c] * w;
      a1 += mv0[r][1][c] * w;
      a2 += mv0[r][2][c] * w;
    }
    vh0[r][0][h] = a0; vh0[r][1][h] = a1; vh0[r][2][h] = a2;
    feat0[r][kSI0 + h] = (_Float16)sqrtf(fmaxf(a0 * a0 + a1 * a1 + a2 * a2, kEps));
  }
  __syncthreads();

  // ---- GVP0 scalar GEMM: [16 x 384] x [384 -> 128], relu -> feat1 cols 0..127
  {
    const _Float16* Bb = W0 + (wid * 16) * kK0;
    __builtin_prefetch(Bb, 0, 3);
    f8_t acc = {};
#pragma unroll
    for (int k0 = 0; k0 < kK0; k0 += 32)
      acc = wmma16(fragA(&feat0[0][0], kK0, k0, lane), fragB(Bb, kK0, k0, lane), acc);
    int c = lane & 15, hf = lane >> 4, col = wid * 16 + c;
    float bias = wsb0[col];
#pragma unroll
    for (int i = 0; i < 8; ++i)
      feat1[i + 8 * hf][col] = (_Float16)fmaxf(acc[i] + bias, 0.f);
  }
  // ---- GVP0 vector out (32 ch) + sigmoid(norm) gate -> mv1
  for (int p = tid; p < 16 * kNV; p += 256) {
    int r = p >> 5, j = p & 31;
    float a0 = 0.f, a1 = 0.f, a2 = 0.f;
    for (int h = 0; h < kVI0; ++h) {
      float w = wv0[j * kVI0 + h];
      a0 += vh0[r][0][h] * w; a1 += vh0[r][1][h] * w; a2 += vh0[r][2][h] * w;
    }
    float g = sigmoidf_(sqrtf(fmaxf(a0 * a0 + a1 * a1 + a2 * a2, kEps)));
    mv1[r][0][j] = a0 * g; mv1[r][1][j] = a1 * g; mv1[r][2][j] = a2 * g;
  }
  __syncthreads();

  // ---- GVP1 vector hidden (32x32); vn -> feat1 cols 128..159
  for (int p = tid; p < 16 * kNV; p += 256) {
    int r = p >> 5, h = p & 31;
    float a0 = 0.f, a1 = 0.f, a2 = 0.f;
    for (int c = 0; c < kNV; ++c) {
      float w = wh1[h * kNV + c];
      a0 += mv1[r][0][c] * w; a1 += mv1[r][1][c] * w; a2 += mv1[r][2][c] * w;
    }
    vh1[r][0][h] = a0; vh1[r][1][h] = a1; vh1[r][2][h] = a2;
    feat1[r][kNS + h] = (_Float16)sqrtf(fmaxf(a0 * a0 + a1 * a1 + a2 * a2, kEps));
  }
  __syncthreads();

  // ---- GVP1 scalar GEMM: [16 x 160] -> 128, relu -> feat2 cols 0..127
  {
    const _Float16* Bb = W1 + (wid * 16) * kK1;
    __builtin_prefetch(Bb, 0, 3);
    f8_t acc = {};
#pragma unroll
    for (int k0 = 0; k0 < kK1; k0 += 32)
      acc = wmma16(fragA(&feat1[0][0], kK1, k0, lane), fragB(Bb, kK1, k0, lane), acc);
    int c = lane & 15, hf = lane >> 4, col = wid * 16 + c;
    float bias = wsb1[col];
#pragma unroll
    for (int i = 0; i < 8; ++i)
      feat2[i + 8 * hf][col] = (_Float16)fmaxf(acc[i] + bias, 0.f);
  }
  // ---- GVP1 vector out + gate -> mv2
  for (int p = tid; p < 16 * kNV; p += 256) {
    int r = p >> 5, j = p & 31;
    float a0 = 0.f, a1 = 0.f, a2 = 0.f;
    for (int h = 0; h < kNV; ++h) {
      float w = wv1[j * kNV + h];
      a0 += vh1[r][0][h] * w; a1 += vh1[r][1][h] * w; a2 += vh1[r][2][h] * w;
    }
    float g = sigmoidf_(sqrtf(fmaxf(a0 * a0 + a1 * a1 + a2 * a2, kEps)));
    mv2[r][0][j] = a0 * g; mv2[r][1][j] = a1 * g; mv2[r][2][j] = a2 * g;
  }
  __syncthreads();

  // ---- GVP2 vector hidden; vn -> feat2 cols 128..159
  for (int p = tid; p < 16 * kNV; p += 256) {
    int r = p >> 5, h = p & 31;
    float a0 = 0.f, a1 = 0.f, a2 = 0.f;
    for (int c = 0; c < kNV; ++c) {
      float w = wh2[h * kNV + c];
      a0 += mv2[r][0][c] * w; a1 += mv2[r][1][c] * w; a2 += mv2[r][2][c] * w;
    }
    vh2[r][0][h] = a0; vh2[r][1][h] = a1; vh2[r][2][h] = a2;
    feat2[r][kNS + h] = (_Float16)sqrtf(fmaxf(a0 * a0 + a1 * a1 + a2 * a2, kEps));
  }
  __syncthreads();

  // ---- GVP2 scalar GEMM (no act) -> atomic add into agg_s[dst]
  {
    const _Float16* Bb = W2 + (wid * 16) * kK1;
    __builtin_prefetch(Bb, 0, 3);
    f8_t acc = {};
#pragma unroll
    for (int k0 = 0; k0 < kK1; k0 += 32)
      acc = wmma16(fragA(&feat2[0][0], kK1, k0, lane), fragB(Bb, kK1, k0, lane), acc);
    int c = lane & 15, hf = lane >> 4, col = wid * 16 + c;
    float bias = wsb2[col];
#pragma unroll
    for (int i = 0; i < 8; ++i) {
      int row = i + 8 * hf;
      atomicAdd(&agg_s[didx[row] * kNS + col], acc[i] + bias);
    }
  }
  // ---- GVP2 vector out (no act) -> atomic add into agg_v[dst]
  for (int p = tid; p < 16 * kNV; p += 256) {
    int r = p >> 5, j = p & 31;
    float a0 = 0.f, a1 = 0.f, a2 = 0.f;
    for (int h = 0; h < kNV; ++h) {
      float w = wv2[j * kNV + h];
      a0 += vh2[r][0][h] * w; a1 += vh2[r][1][h] * w; a2 += vh2[r][2][h] * w;
    }
    int base = (didx[r] * kNV + j) * 3;
    atomicAdd(&agg_v[base + 0], a0);
    atomicAdd(&agg_v[base + 1], a1);
    atomicAdd(&agg_v[base + 2], a2);
  }
  if (tid < 16) atomicAdd(&cnt[didx[tid]], 1.f);
}

// ---------------- kernel 2: node residual + LN + feedforward + LN -----------
__global__ __launch_bounds__(256) void node_update_kernel(
    const float* __restrict__ node_s, const float* __restrict__ node_v,
    const float* __restrict__ agg_s, const float* __restrict__ agg_v,
    const float* __restrict__ cnt, const float* __restrict__ ln0_g,
    const float* __restrict__ ln0_b, const float* __restrict__ ff0_wh,
    const float* __restrict__ ff0_wsb, const float* __restrict__ ff0_wv,
    const float* __restrict__ ff1_wh, const float* __restrict__ ff1_wsb,
    const float* __restrict__ ff1_wv, const float* __restrict__ ln1_g,
    const float* __restrict__ ln1_b, const _Float16* __restrict__ F0,
    const _Float16* __restrict__ F1, float* __restrict__ out_s,
    float* __restrict__ out_v) {
  __shared__ float sbuf[16][kNS];   // post-LN0 scalar
  __shared__ float vbuf[16][3][33]; // post-LN0 vector
  __shared__ __align__(16) _Float16 featA[16][kKF0];
  __shared__ __align__(16) _Float16 featB[16][kKF1];
  __shared__ float vhA[16][3][66];  // ff0 vector hidden (64)
  __shared__ float mvh[16][3][66];  // ff0 vector out (64, gated)
  __shared__ float vhB[16][3][66];  // ff1 vector hidden (64)
  __shared__ float s2[16][kNS];
  __shared__ float v2[16][3][33];
  __shared__ float smu[16], srs[16], svs[16];

  const int tid  = threadIdx.x;
  const int lane = tid & 31;
  const int wid  = tid >> 5;
  const int n0   = blockIdx.x * 16;

  // residual + mean aggregation
  for (int t = tid; t < 16 * kNS; t += 256) {
    int r = t / kNS, c = t % kNS;
    float den = fmaxf(cnt[n0 + r], 1.f);
    sbuf[r][c] = node_s[(n0 + r) * kNS + c] + agg_s[(n0 + r) * kNS + c] / den;
  }
  for (int t = tid; t < 16 * kNV * 3; t += 256) {
    int r = t / (kNV * 3), rem = t % (kNV * 3), c = rem / 3, d = rem % 3;
    float den = fmaxf(cnt[n0 + r], 1.f);
    int gi = ((n0 + r) * kNV + c) * 3 + d;
    vbuf[r][d][c] = node_v[gi] + agg_v[gi] / den;
  }
  __syncthreads();
  // LN0 statistics
  if (tid < 16) {
    float mu = 0.f;
    for (int c = 0; c < kNS; ++c) mu += sbuf[tid][c];
    mu /= kNS;
    float var = 0.f;
    for (int c = 0; c < kNS; ++c) { float d = sbuf[tid][c] - mu; var += d * d; }
    var /= kNS;
    smu[tid] = mu;
    srs[tid] = rsqrtf(var + kLnEps);
    float vs = 0.f;
    for (int c = 0; c < kNV; ++c) {
      float n2 = vbuf[tid][0][c] * vbuf[tid][0][c] +
                 vbuf[tid][1][c] * vbuf[tid][1][c] +
                 vbuf[tid][2][c] * vbuf[tid][2][c];
      vs += fmaxf(n2, kEps);
    }
    svs[tid] = rsqrtf(vs / kNV);
  }
  __syncthreads();
  // apply LN0
  for (int t = tid; t < 16 * kNS; t += 256) {
    int r = t / kNS, c = t % kNS;
    float x = (sbuf[r][c] - smu[r]) * srs[r] * ln0_g[c] + ln0_b[c];
    sbuf[r][c] = x;
    featA[r][c] = (_Float16)x;
  }
  for (int t = tid; t < 16 * kNV * 3; t += 256) {
    int r = t / (kNV * 3), rem = t % (kNV * 3), c = rem / 3, d = rem % 3;
    vbuf[r][d][c] *= svs[r];
  }
  __syncthreads();

  // ---- ff0 vector hidden (64 from 32); vn -> featA cols 128..191
  for (int p = tid; p < 16 * kHV; p += 256) {
    int r = p >> 6, h = p & 63;
    float a0 = 0.f, a1 = 0.f, a2 = 0.f;
    for (int c = 0; c < kNV; ++c) {
      float w = ff0_wh[h * kNV + c];
      a0 += vbuf[r][0][c] * w; a1 += vbuf[r][1][c] * w; a2 += vbuf[r][2][c] * w;
    }
    vhA[r][0][h] = a0; vhA[r][1][h] = a1; vhA[r][2][h] = a2;
    featA[r][kNS + h] = (_Float16)sqrtf(fmaxf(a0 * a0 + a1 * a1 + a2 * a2, kEps));
  }
  __syncthreads();

  // ---- ff0 scalar GEMM: [16 x 192] -> 512 (4 n-tiles/wave), relu -> featB
#pragma unroll
  for (int t4 = 0; t4 < 4; ++t4) {
    int col16 = (wid * 4 + t4) * 16;
    const _Float16* Bb = F0 + col16 * kKF0;
    __builtin_prefetch(Bb, 0, 3);
    f8_t acc = {};
#pragma unroll
    for (int k0 = 0; k0 < kKF0; k0 += 32)
      acc = wmma16(fragA(&featA[0][0], kKF0, k0, lane), fragB(Bb, kKF0, k0, lane), acc);
    int c = lane & 15, hf = lane >> 4, col = col16 + c;
    float bias = ff0_wsb[col];
#pragma unroll
    for (int i = 0; i < 8; ++i)
      featB[i + 8 * hf][col] = (_Float16)fmaxf(acc[i] + bias, 0.f);
  }
  // ---- ff0 vector out (64) + gate -> mvh
  for (int p = tid; p < 16 * kHV; p += 256) {
    int r = p >> 6, j = p & 63;
    float a0 = 0.f, a1 = 0.f, a2 = 0.f;
    for (int h = 0; h < kHV; ++h) {
      float w = ff0_wv[j * kHV + h];
      a0 += vhA[r][0][h] * w; a1 += vhA[r][1][h] * w; a2 += vhA[r][2][h] * w;
    }
    float g = sigmoidf_(sqrtf(fmaxf(a0 * a0 + a1 * a1 + a2 * a2, kEps)));
    mvh[r][0][j] = a0 * g; mvh[r][1][j] = a1 * g; mvh[r][2][j] = a2 * g;
  }
  __syncthreads();

  // ---- ff1 vector hidden (64 from 64); vn -> featB cols 512..575
  for (int p = tid; p < 16 * kHV; p += 256) {
    int r = p >> 6, h = p & 63;
    float a0 = 0.f, a1 = 0.f, a2 = 0.f;
    for (int c = 0; c < kHV; ++c) {
      float w = ff1_wh[h * kHV + c];
      a0 += mvh[r][0][c] * w; a1 += mvh[r][1][c] * w; a2 += mvh[r][2][c] * w;
    }
    vhB[r][0][h] = a0; vhB[r][1][h] = a1; vhB[r][2][h] = a2;
    featB[r][kHS + h] = (_Float16)sqrtf(fmaxf(a0 * a0 + a1 * a1 + a2 * a2, kEps));
  }
  __syncthreads();

  // ---- ff1 scalar GEMM: [16 x 576] -> 128 (no act) + residual -> s2
  {
    const _Float16* Bb = F1 + (wid * 16) * kKF1;
    __builtin_prefetch(Bb, 0, 3);
    f8_t acc = {};
#pragma unroll
    for (int k0 = 0; k0 < kKF1; k0 += 32)
      acc = wmma16(fragA(&featB[0][0], kKF1, k0, lane), fragB(Bb, kKF1, k0, lane), acc);
    int c = lane & 15, hf = lane >> 4, col = wid * 16 + c;
    float bias = ff1_wsb[col];
#pragma unroll
    for (int i = 0; i < 8; ++i) {
      int row = i + 8 * hf;
      s2[row][col] = sbuf[row][col] + acc[i] + bias;
    }
  }
  // ---- ff1 vector out (32, no act) + residual -> v2
  for (int p = tid; p < 16 * kNV; p += 256) {
    int r = p >> 5, j = p & 31;
    float a0 = 0.f, a1 = 0.f, a2 = 0.f;
    for (int h = 0; h < kHV; ++h) {
      float w = ff1_wv[j * kHV + h];
      a0 += vhB[r][0][h] * w; a1 += vhB[r][1][h] * w; a2 += vhB[r][2][h] * w;
    }
    v2[r][0][j] = vbuf[r][0][j] + a0;
    v2[r][1][j] = vbuf[r][1][j] + a1;
    v2[r][2][j] = vbuf[r][2][j] + a2;
  }
  __syncthreads();
  // LN1 statistics
  if (tid < 16) {
    float mu = 0.f;
    for (int c = 0; c < kNS; ++c) mu += s2[tid][c];
    mu /= kNS;
    float var = 0.f;
    for (int c = 0; c < kNS; ++c) { float d = s2[tid][c] - mu; var += d * d; }
    var /= kNS;
    smu[tid] = mu;
    srs[tid] = rsqrtf(var + kLnEps);
    float vs = 0.f;
    for (int c = 0; c < kNV; ++c) {
      float n2 = v2[tid][0][c] * v2[tid][0][c] + v2[tid][1][c] * v2[tid][1][c] +
                 v2[tid][2][c] * v2[tid][2][c];
      vs += fmaxf(n2, kEps);
    }
    svs[tid] = rsqrtf(vs / kNV);
  }
  __syncthreads();
  // write outputs
  for (int t = tid; t < 16 * kNS; t += 256) {
    int r = t / kNS, c = t % kNS;
    out_s[(n0 + r) * kNS + c] = (s2[r][c] - smu[r]) * srs[r] * ln1_g[c] + ln1_b[c];
  }
  for (int t = tid; t < 16 * kNV * 3; t += 256) {
    int r = t / (kNV * 3), rem = t % (kNV * 3), c = rem / 3, d = rem % 3;
    out_v[((n0 + r) * kNV + c) * 3 + d] = v2[r][d][c] * svs[r];
  }
}

// ---------------- launcher ----------------
extern "C" void kernel_launch(void* const* d_in, const int* in_sizes, int n_in,
                              void* d_out, int out_size, void* d_ws,
                              size_t ws_size, hipStream_t stream) {
  const float* node_s = (const float*)d_in[0];
  const float* node_v = (const float*)d_in[1];
  const float* edge_s = (const float*)d_in[2];
  const float* edge_v = (const float*)d_in[3];
  const int* edge_index = (const int*)d_in[4];
  const float* msg0_wh = (const float*)d_in[5];
  const float* msg0_wsw = (const float*)d_in[6];
  const float* msg0_wsb = (const float*)d_in[7];
  const float* msg0_wv = (const float*)d_in[8];
  const float* msg1_wh = (const float*)d_in[9];
  const float* msg1_wsw = (const float*)d_in[10];
  const float* msg1_wsb = (const float*)d_in[11];
  const float* msg1_wv = (const float*)d_in[12];
  const float* msg2_wh = (const float*)d_in[13];
  const float* msg2_wsw = (const float*)d_in[14];
  const float* msg2_wsb = (const float*)d_in[15];
  const float* msg2_wv = (const float*)d_in[16];
  const float* ln0_g = (const float*)d_in[17];
  const float* ln0_b = (const float*)d_in[18];
  const float* ff0_wh = (const float*)d_in[19];
  const float* ff0_wsw = (const float*)d_in[20];
  const float* ff0_wsb = (const float*)d_in[21];
  const float* ff0_wv = (const float*)d_in[22];
  const float* ff1_wh = (const float*)d_in[23];
  const float* ff1_wsw = (const float*)d_in[24];
  const float* ff1_wsb = (const float*)d_in[25];
  const float* ff1_wv = (const float*)d_in[26];
  const float* ln1_g = (const float*)d_in[27];
  const float* ln1_b = (const float*)d_in[28];

  const int Nn = in_sizes[0] / kNS;  // 10000
  const int Ne = in_sizes[4] / 2;    // 200000

  // workspace layout: [f16 weights | agg_s | agg_v | cnt]
  char* ws = (char*)d_ws;
  _Float16* Wf16 = (_Float16*)ws;
  float* agg_s = (float*)(ws + (size_t)WTOT * sizeof(_Float16));
  float* agg_v = agg_s + (size_t)Nn * kNS;
  float* cnt = agg_v + (size_t)Nn * kNV * 3;
  hipMemsetAsync(agg_s, 0, (size_t)Nn * (kNS + kNV * 3 + 1) * sizeof(float),
                 stream);

  convert_weights_kernel<<<(WTOT + 255) / 256, 256, 0, stream>>>(
      msg0_wsw, msg1_wsw, msg2_wsw, ff0_wsw, ff1_wsw, Wf16);

  edge_msg_kernel<<<Ne / 16, 256, 0, stream>>>(
      node_s, node_v, edge_s, edge_v, edge_index, Ne, msg0_wh, msg0_wsb,
      msg0_wv, msg1_wh, msg1_wsb, msg1_wv, msg2_wh, msg2_wsb, msg2_wv,
      Wf16 + OFF_W0, Wf16 + OFF_W1, Wf16 + OFF_W2, agg_s, agg_v, cnt);

  float* out_s = (float*)d_out;
  float* out_v = out_s + (size_t)Nn * kNS;
  node_update_kernel<<<Nn / 16, 256, 0, stream>>>(
      node_s, node_v, agg_s, agg_v, cnt, ln0_g, ln0_b, ff0_wh, ff0_wsb, ff0_wv,
      ff1_wh, ff1_wsb, ff1_wv, ln1_g, ln1_b, Wf16 + OFF_F0, Wf16 + OFF_F1,
      out_s, out_v);
}
